// MaskRCNN_13331578487476
// MI455X (gfx1250) — compile-verified, matrix operand order checked by reading, not run on hardware
//
#include <hip/hip_runtime.h>

typedef __attribute__((ext_vector_type(16))) _Float16 v16h;
typedef __attribute__((ext_vector_type(8)))  float    v8f;

#define POOLSZ 7
#define NCLS   81
#define NROI   2000
#define MPAD   2048
#define CF     256
#define XK     (CF*POOLSZ*POOLSZ)   /* 12544 */
#define N34    448                  /* 81 + 324 padded to x64 */
#define NOUT34 405
#define NANCH  261888
#define OUT_LOGITS 1047552          /* 261888*4 */
#define OUT_BBOX   1209552          /* + 2000*81 */

// -------- Tensor Data Mover support (gfx1250) -----------------------------------
#ifndef FORCE_NO_TDM
#if __has_builtin(__builtin_amdgcn_tensor_load_to_lds)
#define USE_TDM 1
#endif
#endif

#ifdef USE_TDM
typedef unsigned int v4u  __attribute__((ext_vector_type(4)));
typedef int          v8i_ __attribute__((ext_vector_type(8)));
typedef int          v4i_ __attribute__((ext_vector_type(4)));
typedef const __attribute__((address_space(3))) void* lds_cvp;

__device__ __forceinline__ unsigned lds_off(const void* p) {
    return (unsigned)(unsigned long long)(lds_cvp)p;
}

// 2-D f16 tile load: tile_d0 halves per row, tile_d1 rows, row stride = strideK
// halves. LDS rows padded: every 64B stored, insert 16B (pad_interval=3,
// pad_amount=3) -> 80B LDS row stride (= 40-half padded layout).
__device__ __forceinline__ void tdm_load_2d(unsigned lds_byte, const void* gaddr,
                                            unsigned tile_d0, unsigned tile_d1,
                                            unsigned long long strideK)
{
    unsigned long long ga = (unsigned long long)gaddr;
    v4u g0;
    g0.x = 1u;                                       // count=1 (valid user D#)
    g0.y = lds_byte;                                 // lds_addr
    g0.z = (unsigned)ga;                             // global_addr[31:0]
    g0.w = (unsigned)((ga >> 32) & 0x01FFFFFFu) | (2u << 30);  // addr[56:32], type=2
    v8i_ g1;
    g1[0] = (int)((1u<<16) | (1u<<20) | (3u<<22) | (3u<<25));  // 2B elems, pad on, 64B/16B
    g1[1] = (int)((tile_d0 & 0xFFFFu) << 16);        // tensor_dim0[15:0]
    g1[2] = (int)((tile_d0 >> 16) | ((tile_d1 & 0xFFFFu) << 16));
    g1[3] = (int)((tile_d1 >> 16) | (tile_d0 << 16));// tensor_dim1 hi | tile_dim0
    g1[4] = (int)(tile_d1 & 0xFFFFu);                // tile_dim1 (tile_dim2 = 0)
    g1[5] = (int)(unsigned)(strideK & 0xFFFFFFFFu);  // tensor_dim0_stride lo
    g1[6] = (int)(unsigned)((strideK >> 32) & 0xFFFFu);
    g1[7] = 0;
    v4i_ z4 = {0, 0, 0, 0};
    v8i_ z8 = {0, 0, 0, 0, 0, 0, 0, 0};
    __builtin_amdgcn_tensor_load_to_lds(g0, g1, z4, z4, z8, 0);
}
#endif

// ---------------- 1. RPN proposals (elementwise, bandwidth-bound) ----------------
__global__ void proposals_kernel(const float* __restrict__ anchors,
                                 const float* __restrict__ deltas,
                                 float* __restrict__ out)
{
    int a = blockIdx.x * blockDim.x + threadIdx.x;
    if (a >= NANCH) return;
    float a0 = anchors[a*4+0], a1 = anchors[a*4+1], a2 = anchors[a*4+2], a3 = anchors[a*4+3];
    float d0 = deltas[a*4+0],  d1 = deltas[a*4+1],  d2 = deltas[a*4+2],  d3 = deltas[a*4+3];
    float h = a2 - a0, w = a3 - a1;
    float cy = a0 + 0.5f*h + d0*h;
    float cx = a1 + 0.5f*w + d1*w;
    h *= expf(d2); w *= expf(d3);
    float y1 = cy - 0.5f*h, x1 = cx - 0.5f*w;
    float y2 = y1 + h,      x2 = x1 + w;
    y1 = fminf(fmaxf(y1, 0.f), 1.f); x1 = fminf(fmaxf(x1, 0.f), 1.f);
    y2 = fminf(fmaxf(y2, 0.f), 1.f); x2 = fminf(fmaxf(x2, 0.f), 1.f);
    float4 r = make_float4(x1, y1, x2, y2);
    *(float4*)&out[(size_t)a*4] = r;
}

// ---------------- 2. ROIAlign: one block per ROI, single FPN level, f16 output ---
__global__ void roialign_kernel(const float* __restrict__ p2, const float* __restrict__ p3,
                                const float* __restrict__ p4, const float* __restrict__ p5,
                                const float* __restrict__ rois,
                                _Float16* __restrict__ Xh)
{
    int n = blockIdx.x;
    int t = threadIdx.x;
    _Float16* xrow = Xh + (size_t)n * XK;
    if (n >= NROI) {  // zero pad rows 2000..2047
        for (int idx = t; idx < XK; idx += 256) xrow[idx] = (_Float16)0.0f;
        return;
    }
    float rx1 = rois[n*4+0], ry1 = rois[n*4+1], rx2 = rois[n*4+2], ry2 = rois[n*4+3];
    float area = (ry2 - ry1) * (rx2 - rx1);
    int lvl = (int)fminf(fmaxf(rintf(log2f(sqrtf(area) * (1.0f/224.0f))) + 4.0f, 2.0f), 5.0f);
    const float* f; int H;
    if      (lvl == 2) { f = p2; H = 256; }
    else if (lvl == 3) { f = p3; H = 128; }
    else if (lvl == 4) { f = p4; H = 64;  }
    else               { f = p5; H = 32;  }
    float ny1 = ry1*(1.f/1024.f), nx1 = rx1*(1.f/1024.f);
    float ny2 = ry2*(1.f/1024.f), nx2 = rx2*(1.f/1024.f);

    __shared__ int   sy0[POOLSZ], sy1[POOLSZ], sx0[POOLSZ], sx1[POOLSZ];
    __shared__ float sfy[POOLSZ], sfx[POOLSZ];
    if (t < POOLSZ) {
        float lin = (float)t * (1.0f/6.0f);
        float y = (ny1 + lin*(ny2 - ny1)) * (float)(H - 1);
        float x = (nx1 + lin*(nx2 - nx1)) * (float)(H - 1);
        float yf = floorf(y), xf = floorf(x);
        sfy[t] = y - yf; sfx[t] = x - xf;
        int yi = (int)yf, xi = (int)xf;
        sy0[t] = min(max(yi,     0), H-1);
        sy1[t] = min(max(yi + 1, 0), H-1);
        sx0[t] = min(max(xi,     0), H-1);
        sx1[t] = min(max(xi + 1, 0), H-1);
    }
    __syncthreads();

    for (int idx = t; idx < XK; idx += 256) {
        int c  = idx / 49;
        int s  = idx - c*49;
        int py = s / 7, px = s - py*7;
        const float* fc = f + (size_t)c * H * H;
        int y0 = sy0[py], y1i = sy1[py], x0 = sx0[px], x1i = sx1[px];
        float fy = sfy[py], fx = sfx[px];
        float v00 = fc[y0*H  + x0], v01 = fc[y0*H  + x1i];
        float v10 = fc[y1i*H + x0], v11 = fc[y1i*H + x1i];
        float val = v00*(1.f-fy)*(1.f-fx) + v01*(1.f-fy)*fx
                  + v10*fy*(1.f-fx)       + v11*fy*fx;
        xrow[idx] = (_Float16)val;
    }
}

// ---------------- 3. fp32 [K][N] -> f16 transposed [N][K] (LDS-tiled) -----------
__global__ void transpose_cvt_kernel(const float* __restrict__ W, _Float16* __restrict__ WT,
                                     int K, int N)
{
    __shared__ float tile[32][33];
    int k0 = blockIdx.y * 32, n0 = blockIdx.x * 32;
    int tx = threadIdx.x, ty = threadIdx.y;   // 32 x 8
    #pragma unroll
    for (int j = 0; j < 32; j += 8)
        tile[ty+j][tx] = W[(size_t)(k0+ty+j)*N + n0+tx];
    __syncthreads();
    #pragma unroll
    for (int j = 0; j < 32; j += 8)
        WT[(size_t)(n0+ty+j)*K + k0+tx] = (_Float16)tile[tx][ty+j];
}

// ---- fuse w_cls (1024x81) + w_bbox (1024x324) -> f16 W34^T [448][1024], pad 0 ---
__global__ void build_w34_kernel(const float* __restrict__ wcls, const float* __restrict__ wbbox,
                                 _Float16* __restrict__ W34T)
{
    int idx = blockIdx.x * 256 + threadIdx.x;
    if (idx >= N34 * 1024) return;
    int n = idx >> 10, k = idx & 1023;
    float v = 0.f;
    if      (n < NCLS)   v = wcls [k*NCLS     + n];
    else if (n < NOUT34) v = wbbox[k*(NCLS*4) + (n - NCLS)];
    W34T[idx] = (_Float16)v;
}

__global__ void build_bias34_kernel(const float* __restrict__ bcls,
                                    const float* __restrict__ bbbox,
                                    float* __restrict__ b34)
{
    int i = blockIdx.x * 256 + threadIdx.x;
    if (i >= N34) return;
    float v = 0.f;
    if      (i < NCLS)   v = bcls[i];
    else if (i < NOUT34) v = bbbox[i - NCLS];
    b34[i] = v;
}

// ---------------- 4. WMMA GEMM: C = act(A[M,K] * B^T[N,K]^T + bias) --------------
// Block tile 128x64, 8 waves (wave32) in a 4x2 grid, each wave 32x32 = 4 WMMA accs.
// LDS staged by the Tensor Data Mover (wave 0 issues 2-D tile descriptors, TDM
// applies the 40-half row padding), double-buffered so the DMA overlaps compute.
template<bool RELU, bool OUTF16>
__global__ __launch_bounds__(256)
void gemm_wmma_kernel(const _Float16* __restrict__ A, const _Float16* __restrict__ BT,
                      const float* __restrict__ bias, void* __restrict__ Out,
                      int N, int K)
{
    constexpr int BM = 128, BN = 64, BK = 32, LDT = 40;
    __shared__ _Float16 sA[2][BM * LDT];
    __shared__ _Float16 sB[2][BN * LDT];

    const int tid  = threadIdx.x;
    const int lane = tid & 31;
    const int wave = tid >> 5;
    const int wm   = (wave >> 1) * 32;   // wave M offset (0,32,64,96)
    const int wn   = (wave & 1)  * 32;   // wave N offset (0,32)
    const int sel  = lane >> 4;          // half-wave selector
    const int rrow = lane & 15;
    const int m0   = blockIdx.y * BM;
    const int n0   = blockIdx.x * BN;
    const int nk   = K / BK;

    v8f acc[2][2] = {};
    union Frag { v16h v; uint4 q[2]; };

#ifdef USE_TDM
    if (wave == 0) {   // prologue: DMA tile 0 into buffer 0
        tdm_load_2d(lds_off(&sA[0][0]), A  + (size_t)m0*K, BK, BM, (unsigned long long)K);
        tdm_load_2d(lds_off(&sB[0][0]), BT + (size_t)n0*K, BK, BN, (unsigned long long)K);
    }
#endif

    for (int kt = 0; kt < nk; ++kt) {
        const int cur = kt & 1;
#ifdef USE_TDM
        if (wave == 0) {
            if (kt + 1 < nk) {   // DMA next tile into the other buffer
                int k1 = (kt + 1) * BK;
                tdm_load_2d(lds_off(&sA[1-cur][0]), A  + (size_t)m0*K + k1, BK, BM, (unsigned long long)K);
                tdm_load_2d(lds_off(&sB[1-cur][0]), BT + (size_t)n0*K + k1, BK, BN, (unsigned long long)K);
                __builtin_amdgcn_s_wait_tensorcnt(2);   // current tile's pair done
            } else {
                __builtin_amdgcn_s_wait_tensorcnt(0);
            }
        }
#else
        {   // fallback: manual staging of tile kt into buf[cur]
            const int k0 = kt * BK;
            #pragma unroll
            for (int i = 0; i < 2; i++) {
                int idx = tid + i*256;
                int row = idx >> 2, ch = idx & 3;
                *(uint4*)&sA[cur][row*LDT + ch*8] =
                    *(const uint4*)&A[(size_t)(m0+row)*K + k0 + ch*8];
            }
            int row = tid >> 2, ch = tid & 3;
            *(uint4*)&sB[cur][row*LDT + ch*8] =
                *(const uint4*)&BT[(size_t)(n0+row)*K + k0 + ch*8];
            if (k0 + BK < K) {
                __builtin_prefetch(&A [(size_t)(m0 + (tid>>2))*K + k0 + BK + (tid&3)*8], 0, 0);
                __builtin_prefetch(&BT[(size_t)(n0 + ((tid>>2)&63))*K + k0 + BK + (tid&3)*8], 0, 0);
            }
        }
#endif
        __syncthreads();   // buf[cur] ready for all waves

        const _Float16* curA = sA[cur];
        const _Float16* curB = sB[cur];

        // fragment loads: ISA 16-bit A layout -> two contiguous 16B chunks per lane
        Frag a[2], b[2];
        #pragma unroll
        for (int mt = 0; mt < 2; mt++) {
            int row = wm + mt*16 + rrow;
            a[mt].q[0] = *(const uint4*)&curA[row*LDT +      sel*8];
            a[mt].q[1] = *(const uint4*)&curA[row*LDT + 16 + sel*8];
        }
        #pragma unroll
        for (int nt = 0; nt < 2; nt++) {
            int row = wn + nt*16 + rrow;
            b[nt].q[0] = *(const uint4*)&curB[row*LDT +      sel*8];
            b[nt].q[1] = *(const uint4*)&curB[row*LDT + 16 + sel*8];
        }
        #pragma unroll
        for (int mt = 0; mt < 2; mt++)
            #pragma unroll
            for (int nt = 0; nt < 2; nt++)
                acc[mt][nt] = __builtin_amdgcn_wmma_f32_16x16x32_f16(
                    false, a[mt].v, false, b[nt].v, (short)0, acc[mt][nt], false, false);

        __syncthreads();   // all reads of buf[cur] done before it is re-filled
    }

    // epilogue: D layout m = r + 8*sel, n = lane&15
    #pragma unroll
    for (int mt = 0; mt < 2; mt++) {
        #pragma unroll
        for (int nt = 0; nt < 2; nt++) {
            int n = n0 + wn + nt*16 + rrow;
            float bv = bias[n];
            #pragma unroll
            for (int r = 0; r < 8; r++) {
                int m = m0 + wm + mt*16 + sel*8 + r;
                float val = acc[mt][nt][r] + bv;
                if (RELU) val = fmaxf(val, 0.0f);
                if (OUTF16) ((_Float16*)Out)[(size_t)m*N + n] = (_Float16)val;
                else        ((float*)   Out)[(size_t)m*N + n] = val;
            }
        }
    }
}

// ---------------- 5. scatter fused head output to logits / bbox -----------------
__global__ void scatter_kernel(const float* __restrict__ Hout, float* __restrict__ out)
{
    int idx = blockIdx.x * 256 + threadIdx.x;
    if (idx >= NROI * NOUT34) return;
    int n = idx / NOUT34;
    int j = idx - n * NOUT34;
    float v = Hout[(size_t)n * N34 + j];
    if (j < NCLS) out[OUT_LOGITS + n*NCLS     + j]          = v;
    else          out[OUT_BBOX   + n*(NCLS*4) + (j - NCLS)] = v;
}

// ================================ launcher ======================================
extern "C" void kernel_launch(void* const* d_in, const int* in_sizes, int n_in,
                              void* d_out, int out_size, void* d_ws, size_t ws_size,
                              hipStream_t stream)
{
    const float* p2      = (const float*)d_in[0];
    const float* p3      = (const float*)d_in[1];
    const float* p4      = (const float*)d_in[2];
    const float* p5      = (const float*)d_in[3];
    const float* rois    = (const float*)d_in[4];
    const float* anchors = (const float*)d_in[5];
    const float* deltas  = (const float*)d_in[6];
    const float* w_fc1   = (const float*)d_in[7];
    const float* b_fc1   = (const float*)d_in[8];
    const float* w_fc2   = (const float*)d_in[9];
    const float* b_fc2   = (const float*)d_in[10];
    const float* w_cls   = (const float*)d_in[11];
    const float* b_cls   = (const float*)d_in[12];
    const float* w_bbox  = (const float*)d_in[13];
    const float* b_bbox  = (const float*)d_in[14];
    float* out = (float*)d_out;

    char* ws = (char*)d_ws;
    size_t off = 0;
    auto walloc = [&](size_t bytes) -> void* {
        void* p = ws + off;
        off = (off + bytes + 255) & ~(size_t)255;
        return p;
    };
    _Float16* Xh   = (_Float16*)walloc((size_t)MPAD * XK   * 2);  // 51.4 MB
    _Float16* W1T  = (_Float16*)walloc((size_t)XK   * 1024 * 2);  // 25.7 MB
    _Float16* H1h  = (_Float16*)walloc((size_t)MPAD * 1024 * 2);  //  4.2 MB
    _Float16* W2T  = (_Float16*)walloc((size_t)1024 * 1024 * 2);  //  2.1 MB
    _Float16* H2h  = (_Float16*)walloc((size_t)MPAD * 1024 * 2);  //  4.2 MB
    _Float16* W34T = (_Float16*)walloc((size_t)N34  * 1024 * 2);  //  0.9 MB
    float*    b34  = (float*)   walloc((size_t)N34 * 4);
    float*    Hout = (float*)   walloc((size_t)MPAD * N34 * 4);   //  3.7 MB

    // phase 1: proposals -> d_out[0 .. 4A)
    proposals_kernel<<<NANCH/256, 256, 0, stream>>>(anchors, deltas, out);

    // phase 2: ROIAlign -> f16 X [2048][12544]
    roialign_kernel<<<MPAD, 256, 0, stream>>>(p2, p3, p4, p5, rois, Xh);

    // phase 3: weight conversions (fp32 -> transposed f16)
    transpose_cvt_kernel<<<dim3(1024/32, XK/32),  dim3(32,8), 0, stream>>>(w_fc1, W1T, XK,   1024);
    transpose_cvt_kernel<<<dim3(1024/32, 1024/32),dim3(32,8), 0, stream>>>(w_fc2, W2T, 1024, 1024);
    build_w34_kernel<<<(N34*1024 + 255)/256, 256, 0, stream>>>(w_cls, w_bbox, W34T);
    build_bias34_kernel<<<2, 256, 0, stream>>>(b_cls, b_bbox, b34);

    // phase 4: WMMA GEMM chain
    gemm_wmma_kernel<true,  true ><<<dim3(1024/64, MPAD/128), 256, 0, stream>>>(Xh,  W1T,  b_fc1, H1h,  1024, XK);
    gemm_wmma_kernel<true,  true ><<<dim3(1024/64, MPAD/128), 256, 0, stream>>>(H1h, W2T,  b_fc2, H2h,  1024, 1024);
    gemm_wmma_kernel<false, false><<<dim3(N34/64,  MPAD/128), 256, 0, stream>>>(H2h, W34T, b34,   Hout, N34,  1024);

    // phase 5: scatter logits + bbox into d_out
    scatter_kernel<<<(NROI*NOUT34 + 255)/256, 256, 0, stream>>>(Hout, out);
}
